// BiSeqVLADModule_274877907332
// MI455X (gfx1250) — matrix operand order repaired and checked by reading, not compiled, not run on hardware
//
#include <hip/hip_runtime.h>
#include <math.h>

typedef __bf16 bf16_t;
typedef __attribute__((ext_vector_type(16))) __bf16 v16bf;
typedef __attribute__((ext_vector_type(4)))  __bf16 v4bf;
typedef __attribute__((ext_vector_type(8)))  float  v8f;

#define HW 196       // 14*14
#define W14 14
#define TSTEPS 32
#define NFRAMES 256
#define NB 8         // batch
#define KCENT 64
#define DDIM 512
#define CIN 1024

__device__ __forceinline__ bf16_t f2bf(float f) { return (bf16_t)f; }

// ---- CDNA5 WMMA fragment loaders (wave32 layouts, CDNA5 ISA) ----
// Pitch = 40 elements (80 B) -> rows 16B-aligned, ds_load_b128-friendly,
// 20-bank row stride distributes 16 rows across all 64 banks.
// A 16x32 bf16, LDS row-major [m][k]:
// lanes 0-15: m=lane, K 0..15 ; lanes 16-31: m=lane-16, K 16..31
__device__ __forceinline__ v16bf frag_a_lds(const bf16_t* base, int lane, int pitch) {
  int m  = lane & 15;
  int kb = (lane & 16) ? 16 : 0;
  const bf16_t* p0 = base + m * pitch + kb;
  v16bf a;
#pragma unroll
  for (int i = 0; i < 16; ++i) a[i] = p0[i];
  return a;
}
// B 32x16 bf16, LDS stored transposed [n][k]:
// lanes 0-15: n=lane, K 0..15 ; lanes 16-31: n=lane-16, K 16..31
__device__ __forceinline__ v16bf frag_b_lds(const bf16_t* base, int lane, int pitch) {
  int n  = lane & 15;
  int kb = (lane & 16) ? 16 : 0;
  const bf16_t* p = base + n * pitch + kb;
  v16bf b;
#pragma unroll
  for (int i = 0; i < 16; ++i) b[i] = p[i];
  return b;
}

// ---------------------------------------------------------------------------
// GEMM over "pixel layout" activations: A[m,k] = Act[frame(m)][k][pix(m)],
// Bw[n][k] row-major weights, C[m,n] written back in pixel layout + bias.
// 256 threads = 8 waves; wave tile 16(M)x64(N); block MT=WM*16, NT=WN*64.
// Register-staged + double-buffered LDS: global loads for chunk i+1 are in
// flight while chunk i's WMMAs execute; one barrier per K-step.
// ---------------------------------------------------------------------------
template<int WM, int WN>
__global__ void __launch_bounds__(256) gemm_pix_bf16(
    const float* __restrict__ A, const float* __restrict__ Bw,
    const float* __restrict__ bias, float* __restrict__ C,
    int M, int N, int K) {
  constexpr int MT = WM * 16;
  constexpr int NT = WN * 64;
  constexpr int KC = 32;
  constexpr int PI = 40;
  constexpr int EA  = MT * KC / 256;        // scalar A elems / thread (8 or 16)
  constexpr int EB4 = NT * KC / (256 * 4);  // float4 B elems / thread (4 or 2)
  __shared__ alignas(16) bf16_t sA[2][MT * PI];
  __shared__ alignas(16) bf16_t sB[2][NT * PI];

  const int tid  = threadIdx.x;
  const int lane = tid & 31;
  const int wid  = tid >> 5;
  const int wm   = wid % WM;
  const int wn   = wid / WM;
  const int m0   = blockIdx.y * MT;
  const int n0   = blockIdx.x * NT;

  float  ra[EA];
  float4 rb[EB4];

  auto loadA = [&](int k0) {
#pragma unroll
    for (int i = 0; i < EA; ++i) {
      int idx = i * 256 + tid;
      int kk = idx / MT, mi = idx & (MT - 1);
      int mg = m0 + mi, kg = k0 + kk;
      float v = 0.f;
      if (mg < M && kg < K) {
        int fr = mg / HW, p = mg - fr * HW;
        v = A[(size_t)fr * K * HW + (size_t)kg * HW + p];
      }
      ra[i] = v;
    }
  };
  auto loadB = [&](int k0) {
#pragma unroll
    for (int i = 0; i < EB4; ++i) {
      int idx = i * 256 + tid;
      int ni = idx >> 3, k4 = idx & 7;      // KC/4 == 8
      int ng = n0 + ni, kg = k0 + k4 * 4;
      float4 v = make_float4(0.f, 0.f, 0.f, 0.f);
      if (ng < N && kg < K) v = *(const float4*)&Bw[(size_t)ng * K + kg];
      rb[i] = v;
    }
  };
  auto storeAB = [&](int bi) {
#pragma unroll
    for (int i = 0; i < EA; ++i) {
      int idx = i * 256 + tid;
      int kk = idx / MT, mi = idx & (MT - 1);
      sA[bi][mi * PI + kk] = f2bf(ra[i]);
    }
#pragma unroll
    for (int i = 0; i < EB4; ++i) {
      int idx = i * 256 + tid;
      int ni = idx >> 3, kk = (idx & 7) * 4;
      v4bf t; t[0] = f2bf(rb[i].x); t[1] = f2bf(rb[i].y);
              t[2] = f2bf(rb[i].z); t[3] = f2bf(rb[i].w);
      *(v4bf*)&sB[bi][ni * PI + kk] = t;
    }
  };

  v8f zero = {};
  v8f acc[4];
#pragma unroll
  for (int f = 0; f < 4; ++f) acc[f] = zero;

  loadA(0); loadB(0); storeAB(0);
  __syncthreads();
  const int nk = (K + KC - 1) / KC;
  for (int ic = 0; ic < nk; ++ic) {
    const int cur = ic & 1;
    if (ic + 1 < nk) { loadA((ic + 1) * KC); loadB((ic + 1) * KC); }
    v16bf af = frag_a_lds(&sA[cur][wm * 16 * PI], lane, PI);
#pragma unroll
    for (int f = 0; f < 4; ++f) {
      v16bf bf = frag_b_lds(&sB[cur][(wn * 64 + f * 16) * PI], lane, PI);
      acc[f] = __builtin_amdgcn_wmma_f32_16x16x32_bf16(false, af, false, bf,
                                                       (short)0, acc[f], false, false);
    }
    if (ic + 1 < nk) storeAB(cur ^ 1);
    __syncthreads();
  }
  // C layout: VGPR v -> row v (lanes 0-15) / v+8 (lanes 16-31), col = lane&15
#pragma unroll
  for (int f = 0; f < 4; ++f) {
#pragma unroll
    for (int v = 0; v < 8; ++v) {
      int row = m0 + wm * 16 + ((lane & 16) ? v + 8 : v);
      int col = n0 + wn * 64 + f * 16 + (lane & 15);
      if (row < M && col < N) {
        int fr = row / HW, p = row - fr * HW;
        C[(size_t)fr * N * HW + (size_t)col * HW + p] = acc[f][v] + bias[col];
      }
    }
  }
}

// ---------------------------------------------------------------------------
// Batched VLAD GEMM: per frame nt, vladT[nt][k][d] =
//     sum_p assign[nt][k][p]*xr[nt][d][p]  -  asum[nt][k]*centers[k][d]
// M=64 (centers), K=196 pixels (padded to 224), N=512. WM=4, WN=2.
// Both A (assign rows) and B (xr rows) are contiguous along K -> float4 loads.
// ---------------------------------------------------------------------------
__global__ void __launch_bounds__(256) vlad_gemm(
    const float* __restrict__ assign, const float* __restrict__ xr,
    const float* __restrict__ asum, const float* __restrict__ centers,
    float* __restrict__ vladT) {
  constexpr int KC = 32, PI = 40;
  __shared__ alignas(16) bf16_t sA[2][64 * PI];
  __shared__ alignas(16) bf16_t sB[2][128 * PI];

  const int tid  = threadIdx.x;
  const int lane = tid & 31;
  const int wid  = tid >> 5;
  const int wm   = wid % 4;
  const int wn   = wid / 4;
  const int nt   = blockIdx.y;
  const int n0   = blockIdx.x * 128;
  const float* Af = assign + (size_t)nt * KCENT * HW;
  const float* Bf = xr     + (size_t)nt * DDIM * HW;

  float4 ra[2];   // 64*32/(256*4)
  float4 rb[4];   // 128*32/(256*4)

  auto loadA = [&](int k0) {
#pragma unroll
    for (int i = 0; i < 2; ++i) {
      int idx = i * 256 + tid;
      int mi = idx >> 3, kg = k0 + (idx & 7) * 4;
      float4 v = make_float4(0.f, 0.f, 0.f, 0.f);
      if (kg < HW) v = *(const float4*)&Af[(size_t)mi * HW + kg];
      ra[i] = v;
    }
  };
  auto loadB = [&](int k0) {
#pragma unroll
    for (int i = 0; i < 4; ++i) {
      int idx = i * 256 + tid;
      int ni = idx >> 3, kg = k0 + (idx & 7) * 4;
      float4 v = make_float4(0.f, 0.f, 0.f, 0.f);
      if (kg < HW) v = *(const float4*)&Bf[(size_t)(n0 + ni) * HW + kg];
      rb[i] = v;
    }
  };
  auto storeAB = [&](int bi) {
#pragma unroll
    for (int i = 0; i < 2; ++i) {
      int idx = i * 256 + tid;
      int mi = idx >> 3, kk = (idx & 7) * 4;
      v4bf t; t[0] = f2bf(ra[i].x); t[1] = f2bf(ra[i].y);
              t[2] = f2bf(ra[i].z); t[3] = f2bf(ra[i].w);
      *(v4bf*)&sA[bi][mi * PI + kk] = t;
    }
#pragma unroll
    for (int i = 0; i < 4; ++i) {
      int idx = i * 256 + tid;
      int ni = idx >> 3, kk = (idx & 7) * 4;
      v4bf t; t[0] = f2bf(rb[i].x); t[1] = f2bf(rb[i].y);
              t[2] = f2bf(rb[i].z); t[3] = f2bf(rb[i].w);
      *(v4bf*)&sB[bi][ni * PI + kk] = t;
    }
  };

  v8f zero = {};
  v8f acc[4];
#pragma unroll
  for (int f = 0; f < 4; ++f) acc[f] = zero;

  loadA(0); loadB(0); storeAB(0);
  __syncthreads();
  const int nk = 7;                         // 224 / 32
  for (int ic = 0; ic < nk; ++ic) {
    const int cur = ic & 1;
    if (ic + 1 < nk) { loadA((ic + 1) * KC); loadB((ic + 1) * KC); }
    v16bf af = frag_a_lds(&sA[cur][wm * 16 * PI], lane, PI);
#pragma unroll
    for (int f = 0; f < 4; ++f) {
      v16bf bf = frag_b_lds(&sB[cur][(wn * 64 + f * 16) * PI], lane, PI);
      acc[f] = __builtin_amdgcn_wmma_f32_16x16x32_bf16(false, af, false, bf,
                                                       (short)0, acc[f], false, false);
    }
    if (ic + 1 < nk) storeAB(cur ^ 1);
    __syncthreads();
  }
#pragma unroll
  for (int f = 0; f < 4; ++f) {
#pragma unroll
    for (int v = 0; v < 8; ++v) {
      int row = wm * 16 + ((lane & 16) ? v + 8 : v);        // center index k
      int col = n0 + wn * 64 + f * 16 + (lane & 15);        // d
      float val = acc[f][v] - asum[nt * KCENT + row] * centers[(size_t)row * DDIM + col];
      vladT[((size_t)nt * KCENT + row) * DDIM + col] = val;
    }
  }
}

// ---------------------------------------------------------------------------
// GRU step kernels: 3x3 conv on (64ch,14x14), LDS-tiled with halo.
// grid = (2 ktiles of 32, NB batches), block = 256.
// ---------------------------------------------------------------------------
__global__ void gru_zr(const float* __restrict__ hbuf, const float* __restrict__ wxb,
                       const float* __restrict__ Uz, const float* __restrict__ Ur,
                       float* __restrict__ zbuf, float* __restrict__ rhbuf, int t) {
  __shared__ float sh[64 * 16 * 16];
  const int b = blockIdx.y, ktile = blockIdx.x, tid = threadIdx.x;
  for (int i = tid; i < 64 * 256; i += 256) sh[i] = 0.f;
  __syncthreads();
  const float* hb = hbuf + (size_t)b * 64 * HW;
  for (int i = tid; i < 64 * HW; i += 256) {
    int c = i / HW, p = i - c * HW;
    int y = p / W14, x = p - y * W14;
    sh[c * 256 + (y + 1) * 16 + (x + 1)] = hb[i];
  }
  __syncthreads();
  const int nt = b * TSTEPS + t;
  for (int w = tid; w < 32 * HW; w += 256) {
    int kl = w / HW, p = w - kl * HW;
    int k = ktile * 32 + kl;
    int y = p / W14, x = p - y * W14;
    float az = 0.f, ar = 0.f;
    const float* uz = Uz + (size_t)k * 64 * 9;
    const float* ur = Ur + (size_t)k * 64 * 9;
    for (int c = 0; c < 64; ++c) {
      const float* row = sh + c * 256 + y * 16 + x;
#pragma unroll
      for (int dy = 0; dy < 3; ++dy)
#pragma unroll
        for (int dx = 0; dx < 3; ++dx) {
          float hv = row[dy * 16 + dx];
          az = fmaf(hv, uz[c * 9 + dy * 3 + dx], az);
          ar = fmaf(hv, ur[c * 9 + dy * 3 + dx], ar);
        }
    }
    float wt = wxb[((size_t)nt * 64 + k) * HW + p];
    float z = 1.f / (1.f + __expf(-(wt + az)));
    float r = 1.f / (1.f + __expf(-(wt + ar)));
    float hv = sh[k * 256 + (y + 1) * 16 + (x + 1)];
    size_t o = ((size_t)b * 64 + k) * HW + p;
    zbuf[o]  = z;
    rhbuf[o] = r * hv;
  }
}

__global__ void gru_h(const float* __restrict__ rhbuf, const float* __restrict__ wxb,
                      const float* __restrict__ Uh, const float* __restrict__ zbuf,
                      float* __restrict__ hbuf, float* __restrict__ acc_assign, int t) {
  __shared__ float sh[64 * 16 * 16];
  const int b = blockIdx.y, ktile = blockIdx.x, tid = threadIdx.x;
  for (int i = tid; i < 64 * 256; i += 256) sh[i] = 0.f;
  __syncthreads();
  const float* rb = rhbuf + (size_t)b * 64 * HW;
  for (int i = tid; i < 64 * HW; i += 256) {
    int c = i / HW, p = i - c * HW;
    int y = p / W14, x = p - y * W14;
    sh[c * 256 + (y + 1) * 16 + (x + 1)] = rb[i];
  }
  __syncthreads();
  const int nt = b * TSTEPS + t;
  for (int w = tid; w < 32 * HW; w += 256) {
    int kl = w / HW, p = w - kl * HW;
    int k = ktile * 32 + kl;
    int y = p / W14, x = p - y * W14;
    float ah = 0.f;
    const float* uh = Uh + (size_t)k * 64 * 9;
    for (int c = 0; c < 64; ++c) {
      const float* row = sh + c * 256 + y * 16 + x;
#pragma unroll
      for (int dy = 0; dy < 3; ++dy)
#pragma unroll
        for (int dx = 0; dx < 3; ++dx)
          ah = fmaf(row[dy * 16 + dx], uh[c * 9 + dy * 3 + dx], ah);
    }
    float wt = wxb[((size_t)nt * 64 + k) * HW + p];
    float hh = tanhf(wt + ah);
    size_t o = ((size_t)b * 64 + k) * HW + p;
    float z = zbuf[o];
    float hold = hbuf[o];
    float hnew = (1.f - z) * hh + z * hold;
    hbuf[o] = hnew;
    acc_assign[((size_t)nt * 64 + k) * HW + p] += hnew;
  }
}

// ---------------------------------------------------------------------------
__global__ void softmax_k(float* __restrict__ a) {
  int idx = blockIdx.x * 256 + threadIdx.x;   // one pixel per thread
  if (idx >= NFRAMES * HW) return;
  int nt = idx / HW, p = idx - nt * HW;
  float* base = a + (size_t)nt * KCENT * HW + p;
  float mx = -1e30f;
  for (int k = 0; k < KCENT; ++k) mx = fmaxf(mx, base[k * HW]);
  float s = 0.f;
  for (int k = 0; k < KCENT; ++k) { float e = __expf(base[k * HW] - mx); base[k * HW] = e; s += e; }
  float inv = 1.f / s;
  for (int k = 0; k < KCENT; ++k) base[k * HW] *= inv;
}

__global__ void asum_k(const float* __restrict__ a, float* __restrict__ asum) {
  int nt = blockIdx.x, k = threadIdx.x;
  const float* r = a + ((size_t)nt * KCENT + k) * HW;
  float s = 0.f;
  for (int p = 0; p < HW; ++p) s += r[p];
  asum[nt * KCENT + k] = s;
}

__global__ void sum_t(const float* __restrict__ vladT, float* __restrict__ vladB,
                      float* __restrict__ rnorm) {
  int b = blockIdx.x >> 6, k = blockIdx.x & 63;
  __shared__ float red[256];
  float ss = 0.f;
  for (int d = threadIdx.x; d < DDIM; d += 256) {
    float s = 0.f;
    for (int t = 0; t < TSTEPS; ++t)
      s += vladT[(((size_t)(b * TSTEPS + t)) * KCENT + k) * DDIM + d];
    vladB[((size_t)b * KCENT + k) * DDIM + d] = s;
    ss += s * s;
  }
  red[threadIdx.x] = ss;
  __syncthreads();
  for (int off = 128; off > 0; off >>= 1) {
    if (threadIdx.x < off) red[threadIdx.x] += red[threadIdx.x + off];
    __syncthreads();
  }
  if (threadIdx.x == 0) rnorm[b * KCENT + k] = sqrtf(red[0]);
}

__global__ void gnorm_k(const float* __restrict__ rnorm, float* __restrict__ gnorm) {
  int b = blockIdx.x, k = threadIdx.x;
  __shared__ float red[64];
  float rn = rnorm[b * KCENT + k];
  float c = rn / fmaxf(rn, 1e-12f);
  red[k] = c * c;
  __syncthreads();
  for (int off = 32; off > 0; off >>= 1) {
    if (k < off) red[k] += red[k + off];
    __syncthreads();
  }
  if (k == 0) gnorm[b] = sqrtf(red[0]);
}

__global__ void finalize_k(const float* __restrict__ vladB, const float* __restrict__ rnorm,
                           const float* __restrict__ gnorm, float* __restrict__ out) {
  int idx = blockIdx.x * 256 + threadIdx.x;
  if (idx >= NB * KCENT * DDIM) return;
  int b = idx >> 15;           // 64*512
  int k = (idx >> 9) & 63;
  out[idx] = vladB[idx] / (fmaxf(rnorm[b * KCENT + k], 1e-12f) * fmaxf(gnorm[b], 1e-12f));
}

__global__ void zero_k(float* __restrict__ p, int n) {
  int i = blockIdx.x * 256 + threadIdx.x;
  if (i < n) p[i] = 0.f;
}

// ---------------------------------------------------------------------------
extern "C" void kernel_launch(void* const* d_in, const int* in_sizes, int n_in,
                              void* d_out, int out_size, void* d_ws, size_t ws_size,
                              hipStream_t stream) {
  const float* x       = (const float*)d_in[0];
  const float* redu_w  = (const float*)d_in[1];
  const float* redu_b  = (const float*)d_in[2];
  const float* share_w = (const float*)d_in[3];
  const float* share_b = (const float*)d_in[4];
  const float* U_z     = (const float*)d_in[5];
  const float* U_r     = (const float*)d_in[6];
  const float* U_h     = (const float*)d_in[7];
  const float* centers = (const float*)d_in[8];
  float* out = (float*)d_out;
  float* w   = (float*)d_ws;

  float* xr     = w;                                          // 256*512*196
  float* wxb    = xr     + (size_t)NFRAMES * DDIM * HW;       // 256*64*196
  float* assign = wxb    + (size_t)NFRAMES * KCENT * HW;      // 256*64*196
  float* zbuf   = assign + (size_t)NFRAMES * KCENT * HW;      // 8*64*196
  float* rhbuf  = zbuf   + (size_t)NB * KCENT * HW;
  float* hbuf   = rhbuf  + (size_t)NB * KCENT * HW;
  float* asum   = hbuf   + (size_t)NB * KCENT * HW;           // 256*64
  float* vladT  = asum   + (size_t)NFRAMES * KCENT;           // 256*64*512
  float* vladB  = vladT  + (size_t)NFRAMES * KCENT * DDIM;    // 8*64*512
  float* rnorm  = vladB  + (size_t)NB * KCENT * DDIM;         // 8*64
  float* gnormv = rnorm  + (size_t)NB * KCENT;                // 8

  const int M = NFRAMES * HW;  // 50176

  // GEMM1: xr = x * redu_w^T + redu_b   (M x 1024 x 512), 64x128 block tile
  {
    dim3 grid((DDIM + 127) / 128, (M + 63) / 64);
    gemm_pix_bf16<4, 2><<<grid, 256, 0, stream>>>(x, redu_w, redu_b, xr, M, DDIM, CIN);
  }
  // GEMM2: wxb = xr * share_w^T + share_b  (M x 512 x 64), 128x64 block tile
  {
    dim3 grid(1, (M + 127) / 128);
    gemm_pix_bf16<8, 1><<<grid, 256, 0, stream>>>(xr, share_w, share_b, wxb, M, KCENT, DDIM);
  }

  // bidirectional GRU scan
  zero_k<<<(NFRAMES * KCENT * HW + 255) / 256, 256, 0, stream>>>(assign, NFRAMES * KCENT * HW);
  zero_k<<<(NB * KCENT * HW + 255) / 256, 256, 0, stream>>>(hbuf, NB * KCENT * HW);
  for (int t = 0; t < TSTEPS; ++t) {
    gru_zr<<<dim3(2, NB), 256, 0, stream>>>(hbuf, wxb, U_z, U_r, zbuf, rhbuf, t);
    gru_h<<<dim3(2, NB), 256, 0, stream>>>(rhbuf, wxb, U_h, zbuf, hbuf, assign, t);
  }
  zero_k<<<(NB * KCENT * HW + 255) / 256, 256, 0, stream>>>(hbuf, NB * KCENT * HW);
  for (int t = TSTEPS - 1; t >= 0; --t) {
    gru_zr<<<dim3(2, NB), 256, 0, stream>>>(hbuf, wxb, U_z, U_r, zbuf, rhbuf, t);
    gru_h<<<dim3(2, NB), 256, 0, stream>>>(rhbuf, wxb, U_h, zbuf, hbuf, assign, t);
  }

  // softmax over centers, per pixel (in place on assign)
  softmax_k<<<(NFRAMES * HW + 255) / 256, 256, 0, stream>>>(assign);
  asum_k<<<NFRAMES, KCENT, 0, stream>>>(assign, asum);

  // VLAD einsum with fused residual-center subtraction
  vlad_gemm<<<dim3(DDIM / 128, NFRAMES), 256, 0, stream>>>(assign, xr, asum, centers, vladT);

  // temporal sum + intra-normalization + global normalization
  sum_t<<<NB * KCENT, 256, 0, stream>>>(vladT, vladB, rnorm);
  gnorm_k<<<NB, KCENT, 0, stream>>>(rnorm, gnormv);
  finalize_k<<<(NB * KCENT * DDIM + 255) / 256, 256, 0, stream>>>(vladB, rnorm, gnormv, out);
}
